// CMRET_76295799046832
// MI455X (gfx1250) — matrix operand order repaired and checked by reading, not compiled, not run on hardware
//
#include <hip/hip_runtime.h>
#include <math.h>

#define NA     256
#define NM     255
#define NPAIR  (NA * NM)      // 65280
#define FDIM   128
#define F3     384
#define KRBF   50
#define KPAD_E 64
#define NLAYER 4
#define RCUT   5.0f

typedef __attribute__((ext_vector_type(16))) _Float16 v16h;
typedef __attribute__((ext_vector_type(8)))  _Float16 v8h;
typedef __attribute__((ext_vector_type(8)))  float    v8f;

__device__ __forceinline__ float silu_f(float x) { return x / (1.0f + __expf(-x)); }

// ---------------------------------------------------------------------------
// WMMA GEMM: C[M x N] = act(A16[M x Kpad] @ B + bias)
//  - A16: f16 row-major, leading dim = Kpad (multiple of 32, zero padded)
//  - Bt16: f16 COLUMN-major (N x Kpad), i.e. Bt16[n*Kpad + k]
//  - One wave computes a 16x64 tile: one A fragment, 4 B fragments, 4 WMMA.
//  - M % 16 == 0, N % 64 == 0.  act: 0 none, 1 bias+SiLU, 2 bias.
// ---------------------------------------------------------------------------
__global__ void wmma_gemm_f16(const _Float16* __restrict__ A16,
                              const _Float16* __restrict__ Bt16,
                              float* __restrict__ C, int ldc,
                              int M, int N, int Kpad,
                              const float* __restrict__ bias, int act)
{
    const int lane = threadIdx.x & 31;
    const int wave = threadIdx.x >> 5;
    const int tilesN = N >> 6;                       // 64-wide column tiles
    const int tilesTotal = (M >> 4) * tilesN;
    const int tile = blockIdx.x * (blockDim.x >> 5) + wave;
    if (tile >= tilesTotal) return;                  // wave-uniform guard
    const int tm = tile / tilesN;
    const int tn = tile - tm * tilesN;

    const int row16 = lane & 15;
    const int hi    = lane >> 4;                     // lanes 0-15 / 16-31

    const _Float16* Arow = A16 + (size_t)(tm * 16 + row16) * Kpad;
    const _Float16* Bcol0 = Bt16 + (size_t)(tn * 64 + row16) * Kpad;

    v8f acc0 = {}, acc1 = {}, acc2 = {}, acc3 = {};

    for (int kk = 0; kk < Kpad; kk += 32) {
        // A fragment: halves 0..7 = K kk+hi*8.., halves 8..15 = K kk+16+hi*8..
        v8h alo = *(const v8h*)(Arow + kk + hi * 8);
        v8h ahi = *(const v8h*)(Arow + kk + 16 + hi * 8);
        v16h a = __builtin_shufflevector(alo, ahi,
                 0,1,2,3,4,5,6,7,8,9,10,11,12,13,14,15);

        // B fragments: halves 0..7 = K kk+hi*8.., halves 8..15 = K kk+16+hi*8..
        const int kB = kk + hi * 8;
#pragma unroll
        for (int t = 0; t < 4; ++t) {
            const _Float16* bp = Bcol0 + (size_t)(t * 16) * Kpad + kB;
            v8h blo = *(const v8h*)bp;
            v8h bhi = *(const v8h*)(bp + 16);
            v16h b = __builtin_shufflevector(blo, bhi,
                     0,1,2,3,4,5,6,7,8,9,10,11,12,13,14,15);
            if (t == 0) acc0 = __builtin_amdgcn_wmma_f32_16x16x32_f16(false, a, false, b, (short)0, acc0, false, false);
            if (t == 1) acc1 = __builtin_amdgcn_wmma_f32_16x16x32_f16(false, a, false, b, (short)0, acc1, false, false);
            if (t == 2) acc2 = __builtin_amdgcn_wmma_f32_16x16x32_f16(false, a, false, b, (short)0, acc2, false, false);
            if (t == 3) acc3 = __builtin_amdgcn_wmma_f32_16x16x32_f16(false, a, false, b, (short)0, acc3, false, false);
        }
    }

    // C layout: VGPR r -> row tm*16 + r + 8*hi, col = subtile base + row16
    const int crow = tm * 16 + hi * 8;
#pragma unroll
    for (int t = 0; t < 4; ++t) {
        v8f acc = (t == 0) ? acc0 : (t == 1) ? acc1 : (t == 2) ? acc2 : acc3;
        const int ccol = tn * 64 + t * 16 + row16;
        float bb = act ? bias[ccol] : 0.0f;
#pragma unroll
        for (int r = 0; r < 8; ++r) {
            float x = acc[r] + bb;
            if (act == 1) x = silu_f(x);
            C[(size_t)(crow + r) * ldc + ccol] = x;
        }
    }
}

// ---------------------------------------------------------------------------
// f32 (M x K) row-major -> f16 (M x Kpad) row-major, zero padded
// ---------------------------------------------------------------------------
__global__ void pack_a_f16(const float* __restrict__ x, _Float16* __restrict__ y,
                           int M, int K, int Kpad)
{
    int t = blockIdx.x * blockDim.x + threadIdx.x;
    if (t >= M * Kpad) return;
    int r = t / Kpad, k = t - r * Kpad;
    y[t] = (k < K) ? (_Float16)x[(size_t)r * K + k] : (_Float16)0.0f;
}

// ---------------------------------------------------------------------------
// f32 B (K x N) row-major -> f16 Bt (N x Kpad) column-major, zero padded
// ---------------------------------------------------------------------------
__global__ void pack_bt_f16(const float* __restrict__ B, _Float16* __restrict__ yt,
                            int K, int N, int Kpad)
{
    int t = blockIdx.x * blockDim.x + threadIdx.x;
    if (t >= N * Kpad) return;
    int n = t / Kpad, k = t - n * Kpad;
    yt[t] = (k < K) ? (_Float16)B[(size_t)k * N + n] : (_Float16)0.0f;
}

// ---------------------------------------------------------------------------
// Geometry: per edge (i,m): unit vector, cutoff, pair mask, RBF e -> f16 padded
// ---------------------------------------------------------------------------
__global__ void geom_kernel(const float* __restrict__ R,
                            float* __restrict__ vecn, float* __restrict__ cof,
                            float* __restrict__ pmb, _Float16* __restrict__ e16)
{
    int p = blockIdx.x * blockDim.x + threadIdx.x;
    if (p >= NPAIR) return;
    int i = p / NM, m = p - i * NM;
    int j = m + (m >= i ? 1 : 0);
    float pm = ((i >> 5) == (j >> 5)) ? 1.0f : 0.0f;   // 32 atoms / molecule
    float vx = (R[i * 3 + 0] - R[j * 3 + 0]) * pm;
    float vy = (R[i * 3 + 1] - R[j * 3 + 1]) * pm;
    float vz = (R[i * 3 + 2] - R[j * 3 + 2]) * pm;
    float sq = vx * vx + vy * vy + vz * vz;
    float d  = (sq > 0.0f) ? sqrtf(sq) : 0.0f;
    float inv = (d > 0.0f) ? (1.0f / d) : 0.0f;
    vecn[p * 3 + 0] = vx * inv;
    vecn[p * 3 + 1] = vy * inv;
    vecn[p * 3 + 2] = vz * inv;
    float co = 0.5f * (__cosf(3.14159265358979f * d / RCUT) + 1.0f)
             * ((d <= RCUT) ? 1.0f : 0.0f) * pm;
    cof[p] = co;
    pmb[p] = pm;
    const float gamma = 0.5f / ((RCUT / (KRBF - 1)) * (RCUT / (KRBF - 1)));
    const float dmu   = RCUT / (KRBF - 1);
    _Float16* ep = e16 + (size_t)p * KPAD_E;
#pragma unroll 5
    for (int k = 0; k < KRBF; ++k) {
        float dd = d - dmu * (float)k;
        ep[k] = (_Float16)(__expf(-gamma * dd * dd) * co);
    }
#pragma unroll
    for (int k = KRBF; k < KPAD_E; ++k) ep[k] = (_Float16)0.0f;
}

// ---------------------------------------------------------------------------
__global__ void init_kernel(const int* __restrict__ Z, const float* __restrict__ embed,
                            float* __restrict__ s, float* __restrict__ o,
                            float* __restrict__ v)
{
    int t = blockIdx.x * blockDim.x + threadIdx.x;   // NA*FDIM threads
    int i = t >> 7, f = t & 127;
    s[t] = embed[Z[i] * FDIM + f];
    o[t] = 0.0f;
    v[i * F3 + f]            = 0.0f;
    v[i * F3 + FDIM + f]     = 0.0f;
    v[i * F3 + 2 * FDIM + f] = 0.0f;
}

// ---------------------------------------------------------------------------
__global__ void ln_kernel(const float* __restrict__ x, const float* __restrict__ g,
                          const float* __restrict__ b, float* __restrict__ y)
{
    __shared__ float red[FDIM];
    int row = blockIdx.x, f = threadIdx.x;
    float val = x[row * FDIM + f];
    red[f] = val; __syncthreads();
    for (int s = 64; s > 0; s >>= 1) { if (f < s) red[f] += red[f + s]; __syncthreads(); }
    float mu = red[0] * (1.0f / FDIM); __syncthreads();
    float dd = val - mu;
    red[f] = dd * dd; __syncthreads();
    for (int s = 64; s > 0; s >>= 1) { if (f < s) red[f] += red[f + s]; __syncthreads(); }
    float var = red[0] * (1.0f / FDIM);
    y[row * FDIM + f] = dd * rsqrtf(var + 1e-5f) * g[f] + b[f];
}

// ---------------------------------------------------------------------------
__global__ void logits_kernel(const float* __restrict__ q, const float* __restrict__ k,
                              const float* __restrict__ dk, const float* __restrict__ pmb,
                              float* __restrict__ lg)
{
    int t = blockIdx.x * blockDim.x + threadIdx.x;
    if (t >= NPAIR * 4) return;
    int h = t & 3;
    int p = t >> 2;
    int i = p / NM, m = p - i * NM;
    int j = m + (m >= i ? 1 : 0);
    const float* qp  = q  + i * FDIM + h * 32;
    const float* kp  = k  + j * FDIM + h * 32;
    const float* dkp = dk + (size_t)p * FDIM + h * 32;
    float acc = 0.0f;
#pragma unroll
    for (int d = 0; d < 32; ++d) acc += qp[d] * kp[d] * dkp[d];
    const float inv_temp = 0.088388347648318447f;    // 1 / (2*sqrt(32))
    lg[t] = (pmb[p] > 0.0f) ? acc * inv_temp : -1.0e9f;
}

// ---------------------------------------------------------------------------
__global__ void softmax_kernel(const float* __restrict__ lg, const float* __restrict__ cof,
                               float* __restrict__ attn)
{
    __shared__ float red[256];
    int ih = blockIdx.x;
    int i = ih >> 2, h = ih & 3;
    int m = threadIdx.x;
    float x = -1.0e30f;
    if (m < NM) x = lg[((size_t)(i * NM + m)) * 4 + h];
    red[m] = x; __syncthreads();
    for (int s = 128; s > 0; s >>= 1) { if (m < s) red[m] = fmaxf(red[m], red[m + s]); __syncthreads(); }
    float mx = red[0]; __syncthreads();
    float ex = (m < NM) ? __expf(x - mx) : 0.0f;
    red[m] = ex; __syncthreads();
    for (int s = 128; s > 0; s >>= 1) { if (m < s) red[m] += red[m + s]; __syncthreads(); }
    float inv = 1.0f / red[0];
    if (m < NM) attn[((size_t)(i * NM + m)) * 4 + h] = ex * inv * cof[i * NM + m];
}

// ---------------------------------------------------------------------------
__global__ void aggregate_kernel(const float* __restrict__ attn, const float* __restrict__ val,
                                 const float* __restrict__ dv, const float* __restrict__ v,
                                 const float* __restrict__ vecn,
                                 float* __restrict__ dsb, float* __restrict__ dw)
{
    int i = blockIdx.x;
    int f = threadIdx.x;
    int h = f >> 5;
    float accs = 0.0f, a0 = 0.0f, a1 = 0.0f, a2 = 0.0f;
    for (int m = 0; m < NM; ++m) {
        int p = i * NM + m;
        int j = m + (m >= i ? 1 : 0);
        float a = attn[(size_t)p * 4 + h];
        const float* vj  = val + j * F3;
        const float* dvp = dv + (size_t)p * F3;
        float x1 = vj[f]            * dvp[f];
        float x2 = vj[FDIM + f]     * dvp[FDIM + f];
        float x3 = vj[2 * FDIM + f] * dvp[2 * FDIM + f];
        accs += a * x1;
        float n0 = vecn[p * 3 + 0], n1 = vecn[p * 3 + 1], n2 = vecn[p * 3 + 2];
        const float* vv = v + j * F3;
        a0 += a * (x2 * vv[f]            + x3 * n0);
        a1 += a * (x2 * vv[FDIM + f]     + x3 * n1);
        a2 += a * (x2 * vv[2 * FDIM + f] + x3 * n2);
    }
    dsb[i * FDIM + f] = accs;
    dw[i * F3 + f]            = a0;
    dw[i * F3 + FDIM + f]     = a1;
    dw[i * F3 + 2 * FDIM + f] = a2;
}

// ---------------------------------------------------------------------------
__global__ void update_kernel(const float* __restrict__ wo, const float* __restrict__ v1,
                              const float* __restrict__ v2, const float* __restrict__ v3,
                              const float* __restrict__ dw,
                              float* __restrict__ s, float* __restrict__ o,
                              float* __restrict__ v)
{
    int t = blockIdx.x * blockDim.x + threadIdx.x;
    if (t >= NA * FDIM) return;
    int i = t >> 7, f = t & 127;
    float dot = 0.0f;
#pragma unroll
    for (int c = 0; c < 3; ++c)
        dot += v1[i * F3 + c * FDIM + f] * v2[i * F3 + c * FDIM + f];
    float o1 = wo[i * F3 + f];
    float o2 = wo[i * F3 + FDIM + f];
    float o3 = wo[i * F3 + 2 * FDIM + f];
    float dx = o2 + o3 * dot;
    s[t] += dx;
    o[t] += dx;
#pragma unroll
    for (int c = 0; c < 3; ++c) {
        int idx = i * F3 + c * FDIM + f;
        v[idx] += dw[idx] + o1 * v3[idx];
    }
}

// ---------------------------------------------------------------------------
__global__ void out_kernel(const float* __restrict__ h1, const float* __restrict__ w2,
                           const float* __restrict__ b2, const float* __restrict__ batch,
                           float* __restrict__ out)
{
    __shared__ float atom[NA];
    __shared__ float red[NA];
    int i = threadIdx.x;
    float a = b2[0];
#pragma unroll 8
    for (int f = 0; f < 64; ++f) a += h1[i * 64 + f] * w2[f];
    atom[i] = a; __syncthreads();
    for (int b = 0; b < 8; ++b) {
        red[i] = batch[b * NA + i] * atom[i]; __syncthreads();
        for (int s = 128; s > 0; s >>= 1) { if (i < s) red[i] += red[i + s]; __syncthreads(); }
        if (i == 0) out[b] = red[0];
        __syncthreads();
    }
}

// ---------------------------------------------------------------------------
static inline void launch_gemm(const _Float16* A16, const _Float16* Bt16,
                               float* C, int ldc, int M, int N, int Kpad,
                               const float* bias, int act, hipStream_t st)
{
    int tiles  = (M / 16) * (N / 64);
    int blocks = (tiles + 3) / 4;                 // 4 waves (128 threads) / block
    hipLaunchKernelGGL(wmma_gemm_f16, dim3(blocks), dim3(128), 0, st,
                       A16, Bt16, C, ldc, M, N, Kpad, bias, act);
}

static inline void launch_pack_a(const float* x, _Float16* y, int M, int K, int Kpad,
                                 hipStream_t st)
{
    int n = M * Kpad;
    hipLaunchKernelGGL(pack_a_f16, dim3((n + 255) / 256), dim3(256), 0, st, x, y, M, K, Kpad);
}

static inline void launch_pack_bt(const float* B, _Float16* yt, int K, int N, int Kpad,
                                  hipStream_t st)
{
    int n = N * Kpad;
    hipLaunchKernelGGL(pack_bt_f16, dim3((n + 255) / 256), dim3(256), 0, st, B, yt, K, N, Kpad);
}

extern "C" void kernel_launch(void* const* d_in, const int* in_sizes, int n_in,
                              void* d_out, int out_size, void* d_ws, size_t ws_size,
                              hipStream_t stream)
{
    (void)in_sizes; (void)n_in; (void)out_size; (void)ws_size;
    const int*   Z      = (const int*)  d_in[0];
    const float* R      = (const float*)d_in[1];
    const float* batch  = (const float*)d_in[2];
    const float* embed  = (const float*)d_in[3];
    const float* ln_g   = (const float*)d_in[4];
    const float* ln_b   = (const float*)d_in[5];
    const float* Wq     = (const float*)d_in[6];
    const float* Wk     = (const float*)d_in[7];
    const float* Wv     = (const float*)d_in[8];
    const float* Wdk    = (const float*)d_in[9];
    const float* bdk    = (const float*)d_in[10];
    const float* Wdv    = (const float*)d_in[11];
    const float* bdv    = (const float*)d_in[12];
    const float* Wo     = (const float*)d_in[13];
    const float* bo     = (const float*)d_in[14];
    const float* U1     = (const float*)d_in[15];
    const float* U2     = (const float*)d_in[16];
    const float* U3     = (const float*)d_in[17];
    const float* lnf_g  = (const float*)d_in[18];
    const float* lnf_b  = (const float*)d_in[19];
    const float* out_w1 = (const float*)d_in[20];
    const float* out_b1 = (const float*)d_in[21];
    const float* out_w2 = (const float*)d_in[22];
    const float* out_b2 = (const float*)d_in[23];

    char* base = (char*)d_ws;
    size_t off = 0;
    auto allocb = [&](size_t bytes) {
        void* p = base + off;
        off += (bytes + 255) & ~(size_t)255;
        return p;
    };
    auto allocf = [&](size_t n) { return (float*)allocb(n * sizeof(float)); };
    auto alloch = [&](size_t n) { return (_Float16*)allocb(n * sizeof(_Float16)); };

    // f32 buffers
    float* vecn = allocf((size_t)NPAIR * 3);
    float* cof  = allocf(NPAIR);
    float* pmb  = allocf(NPAIR);
    float* s    = allocf(NA * FDIM);
    float* o    = allocf(NA * FDIM);
    float* v    = allocf(NA * F3);
    float* sl   = allocf(NA * FDIM);
    float* q    = allocf(NA * FDIM);
    float* kb   = allocf(NA * FDIM);
    float* val  = allocf(NA * F3);
    float* dk   = allocf((size_t)NPAIR * FDIM);
    float* dv   = allocf((size_t)NPAIR * F3);
    float* lg   = allocf((size_t)NPAIR * 4);
    float* at   = allocf((size_t)NPAIR * 4);
    float* dsb  = allocf(NA * FDIM);
    float* dw   = allocf(NA * F3);
    float* v1   = allocf(NA * F3);
    float* v2   = allocf(NA * F3);
    float* v3   = allocf(NA * F3);
    float* wo_  = allocf(NA * F3);
    float* of   = allocf(NA * FDIM);
    float* h1   = allocf(NA * 64);
    // f16 buffers
    _Float16* e16   = alloch((size_t)NPAIR * KPAD_E);  // RBF, packed once
    _Float16* sl16  = alloch(NA * FDIM);
    _Float16* v16b  = alloch((size_t)NA * 3 * FDIM);
    _Float16* dsb16 = alloch(NA * FDIM);
    _Float16* of16  = alloch(NA * FDIM);
    _Float16* w16   = alloch((size_t)F3 * FDIM);       // reusable B buffer (max 384x128)

    hipLaunchKernelGGL(geom_kernel, dim3((NPAIR + 255) / 256), dim3(256), 0, stream,
                       R, vecn, cof, pmb, e16);
    hipLaunchKernelGGL(init_kernel, dim3((NA * FDIM) / 256), dim3(256), 0, stream,
                       Z, embed, s, o, v);

    for (int l = 0; l < NLAYER; ++l) {
        const float* Wq_l  = Wq  + (size_t)l * FDIM * FDIM;
        const float* Wk_l  = Wk  + (size_t)l * FDIM * FDIM;
        const float* Wv_l  = Wv  + (size_t)l * FDIM * F3;
        const float* Wdk_l = Wdk + (size_t)l * KRBF * FDIM;
        const float* bdk_l = bdk + (size_t)l * FDIM;
        const float* Wdv_l = Wdv + (size_t)l * KRBF * F3;
        const float* bdv_l = bdv + (size_t)l * F3;
        const float* Wo_l  = Wo  + (size_t)l * FDIM * F3;
        const float* bo_l  = bo  + (size_t)l * F3;
        const float* U1_l  = U1  + (size_t)l * FDIM * FDIM;
        const float* U2_l  = U2  + (size_t)l * FDIM * FDIM;
        const float* U3_l  = U3  + (size_t)l * FDIM * FDIM;

        hipLaunchKernelGGL(ln_kernel, dim3(NA), dim3(FDIM), 0, stream,
                           s, ln_g + l * FDIM, ln_b + l * FDIM, sl);
        launch_pack_a(sl, sl16, NA, FDIM, FDIM, stream);

        launch_pack_bt(Wq_l, w16, FDIM, FDIM, FDIM, stream);
        launch_gemm(sl16, w16, q, FDIM, NA, FDIM, FDIM, nullptr, 0, stream);
        launch_pack_bt(Wk_l, w16, FDIM, FDIM, FDIM, stream);
        launch_gemm(sl16, w16, kb, FDIM, NA, FDIM, FDIM, nullptr, 0, stream);
        launch_pack_bt(Wv_l, w16, FDIM, F3, FDIM, stream);
        launch_gemm(sl16, w16, val, F3, NA, F3, FDIM, nullptr, 0, stream);

        // dominant GEMMs: 65280 x {128,384} x 50(->64) with fused bias + SiLU
        launch_pack_bt(Wdk_l, w16, KRBF, FDIM, KPAD_E, stream);
        launch_gemm(e16, w16, dk, FDIM, NPAIR, FDIM, KPAD_E, bdk_l, 1, stream);
        launch_pack_bt(Wdv_l, w16, KRBF, F3, KPAD_E, stream);
        launch_gemm(e16, w16, dv, F3, NPAIR, F3, KPAD_E, bdv_l, 1, stream);

        hipLaunchKernelGGL(logits_kernel, dim3((NPAIR * 4 + 255) / 256), dim3(256), 0, stream,
                           q, kb, dk, pmb, lg);
        hipLaunchKernelGGL(softmax_kernel, dim3(NA * 4), dim3(256), 0, stream,
                           lg, cof, at);
        hipLaunchKernelGGL(aggregate_kernel, dim3(NA), dim3(FDIM), 0, stream,
                           at, val, dv, v, vecn, dsb, dw);

        launch_pack_a(v, v16b, NA * 3, FDIM, FDIM, stream);
        launch_pack_bt(U1_l, w16, FDIM, FDIM, FDIM, stream);
        launch_gemm(v16b, w16, v1, FDIM, NA * 3, FDIM, FDIM, nullptr, 0, stream);
        launch_pack_bt(U2_l, w16, FDIM, FDIM, FDIM, stream);
        launch_gemm(v16b, w16, v2, FDIM, NA * 3, FDIM, FDIM, nullptr, 0, stream);
        launch_pack_bt(U3_l, w16, FDIM, FDIM, FDIM, stream);
        launch_gemm(v16b, w16, v3, FDIM, NA * 3, FDIM, FDIM, nullptr, 0, stream);

        launch_pack_a(dsb, dsb16, NA, FDIM, FDIM, stream);
        launch_pack_bt(Wo_l, w16, FDIM, F3, FDIM, stream);
        launch_gemm(dsb16, w16, wo_, F3, NA, F3, FDIM, bo_l, 2, stream);

        hipLaunchKernelGGL(update_kernel, dim3((NA * FDIM) / 256), dim3(256), 0, stream,
                           wo_, v1, v2, v3, dw, s, o, v);
    }

    hipLaunchKernelGGL(ln_kernel, dim3(NA), dim3(FDIM), 0, stream, o, lnf_g, lnf_b, of);
    launch_pack_a(of, of16, NA, FDIM, FDIM, stream);
    launch_pack_bt(out_w1, w16, FDIM, 64, FDIM, stream);
    launch_gemm(of16, w16, h1, 64, NA, 64, FDIM, out_b1, 1, stream);
    hipLaunchKernelGGL(out_kernel, dim3(1), dim3(NA), 0, stream,
                       h1, out_w2, out_b2, batch, (float*)d_out);
}
